// ieHGCNConv_20401094656593
// MI455X (gfx1250) — matrix operand order repaired and checked by reading, not compiled, not run on hardware
//
#include <hip/hip_runtime.h>
#include <hip/hip_bf16.h>
#include <math.h>

// ---------------------------------------------------------------------------
// ieHGCNConv for MI455X (gfx1250, wave32, WMMA)
//   NA=NP=50000, E=800000, IN=256, OUT=128, ATTN=64
// ---------------------------------------------------------------------------

typedef __attribute__((ext_vector_type(16))) __bf16 v16bf;
typedef __attribute__((ext_vector_type(8)))  __bf16 v8bf;
typedef __attribute__((ext_vector_type(8)))  float  v8f;

#define KDIM 256
#define NDIM 128

// ---------------------------------------------------------------------------
// Fold (z@Wq + bq)@War + bar  ->  z @ vec + scalar     (linear composition)
// Wq: [128,64] row-major, Wr: [64], bq: [64], br: [1]
// ---------------------------------------------------------------------------
__global__ __launch_bounds__(128) void combine_attn(
    const float* __restrict__ Wq, const float* __restrict__ Wr,
    const float* __restrict__ bq, const float* __restrict__ br,
    float* __restrict__ outv, float* __restrict__ outs) {
  int j = threadIdx.x;  // 0..127
  float s = 0.f;
  #pragma unroll 8
  for (int t = 0; t < 64; ++t) s += Wq[j * 64 + t] * Wr[t];
  outv[j] = s;
  if (j == 0) {
    float b = 0.f;
    for (int t = 0; t < 64; ++t) b += bq[t] * Wr[t];
    outs[0] = b + br[0];
  }
}

// ---------------------------------------------------------------------------
// C[M,128] = A[M,256] @ W[256,128]   (fp32 in/out, bf16 WMMA compute)
// Block = 128 threads = 4 waves. Wave w computes rows [ (4*blk+w)*16 , +16 )
// across all 128 columns (8 WMMA N-tiles). W is staged transposed in LDS as
// bf16 so B fragments are two contiguous 16-byte ds loads.
// ---------------------------------------------------------------------------
__global__ __launch_bounds__(128) void gemm_bf16_wmma(
    const float* __restrict__ A, const float* __restrict__ W,
    float* __restrict__ C, int M) {
  __shared__ __bf16 wT[NDIM * KDIM];  // [n][k] transposed, 64 KB

  const int tid = threadIdx.x;
  // stage W (row-major [k][n]) -> wT[n][k] as bf16; coalesced global reads
  for (int i = tid; i < KDIM * NDIM; i += 128) {
    int k = i >> 7;        // i / 128
    int n = i & 127;       // i % 128
    wT[n * KDIM + k] = (__bf16)W[i];
  }
  __syncthreads();

  const int wave = tid >> 5;
  const int lane = tid & 31;
  const int half = lane >> 4;     // 0: lanes 0-15, 1: lanes 16-31
  const int l16  = lane & 15;

  const int row0 = (blockIdx.x * 4 + wave) * 16;
  if (row0 >= M) return;

  // A row this lane feeds (clamped for safety on ragged M)
  int arow_i = row0 + l16;
  if (arow_i >= M) arow_i = M - 1;
  const float* __restrict__ arow = A + (size_t)arow_i * KDIM;

  v8f acc[8];
  #pragma unroll
  for (int nt = 0; nt < 8; ++nt) acc[nt] = (v8f){0.f,0.f,0.f,0.f,0.f,0.f,0.f,0.f};

  for (int k0 = 0; k0 < KDIM; k0 += 32) {
    // --- A fragment: 16x32 bf16. Lane (M=l16, half): elems 0..15 at
    //     K=k0+half*16+i (contiguous, matching B).
    const float4* ap0 = (const float4*)(arow + k0 + half * 16);
    const float4* ap1 = (const float4*)(arow + k0 + half * 16 + 8);
    float4 a0 = ap0[0], a1 = ap0[1];
    float4 a2 = ap1[0], a3 = ap1[1];
    v16bf afrag;
    afrag[0]  = (__bf16)a0.x; afrag[1]  = (__bf16)a0.y;
    afrag[2]  = (__bf16)a0.z; afrag[3]  = (__bf16)a0.w;
    afrag[4]  = (__bf16)a1.x; afrag[5]  = (__bf16)a1.y;
    afrag[6]  = (__bf16)a1.z; afrag[7]  = (__bf16)a1.w;
    afrag[8]  = (__bf16)a2.x; afrag[9]  = (__bf16)a2.y;
    afrag[10] = (__bf16)a2.z; afrag[11] = (__bf16)a2.w;
    afrag[12] = (__bf16)a3.x; afrag[13] = (__bf16)a3.y;
    afrag[14] = (__bf16)a3.z; afrag[15] = (__bf16)a3.w;

    #pragma unroll
    for (int nt = 0; nt < 8; ++nt) {
      // --- B fragment: 32x16 bf16. Lane (N=l16, half): K = k0 + half*16 + i,
      //     contiguous in wT -> two 16B LDS loads.
      const int ncol = nt * 16 + l16;
      const v8bf* bp = (const v8bf*)(wT + ncol * KDIM + k0 + half * 16);
      v8bf blo = bp[0];
      v8bf bhi = bp[1];
      v16bf bfrag = __builtin_shufflevector(blo, bhi,
          0, 1, 2, 3, 4, 5, 6, 7, 8, 9, 10, 11, 12, 13, 14, 15);

      acc[nt] = __builtin_amdgcn_wmma_f32_16x16x32_bf16(
          /*neg_a=*/false, afrag, /*neg_b=*/false, bfrag,
          /*c_mod=*/(short)0, acc[nt], /*reuse_a=*/false, /*reuse_b=*/false);
    }
  }

  // D layout: VGPR r of acc -> M = row0 + r + 8*half, N = nt*16 + l16
  #pragma unroll
  for (int nt = 0; nt < 8; ++nt) {
    const int n = nt * 16 + l16;
    #pragma unroll
    for (int r = 0; r < 8; ++r) {
      const int m = row0 + r + half * 8;
      if (m < M) C[(size_t)m * NDIM + n] = acc[nt][r];
    }
  }
}

// ---------------------------------------------------------------------------
// Edge scatter: one wave per edge. Gather 128-float message m[src], atomic
// f32 add into agg[dst], count degree. agg/deg are L2-resident (25.6 MB).
// ---------------------------------------------------------------------------
__global__ __launch_bounds__(256) void edge_scatter(
    const float* __restrict__ m, const int* __restrict__ src,
    const int* __restrict__ dst, float* __restrict__ agg,
    float* __restrict__ deg, int nedges) {
  const int e = (int)((blockIdx.x * (size_t)blockDim.x + threadIdx.x) >> 5);
  const int lane = threadIdx.x & 31;
  if (e >= nedges) return;
  const int s = src[e];
  const int d = dst[e];
  const float4 v = ((const float4*)(m + (size_t)s * NDIM))[lane];
  float* ad = agg + (size_t)d * NDIM + lane * 4;
  atomicAdd(ad + 0, v.x);
  atomicAdd(ad + 1, v.y);
  atomicAdd(ad + 2, v.z);
  atomicAdd(ad + 3, v.w);
  if (lane == 0) atomicAdd(deg + d, 1.0f);
}

// ---------------------------------------------------------------------------
// Finalize helpers
// ---------------------------------------------------------------------------
__device__ __forceinline__ float wave_sum32(float v) {
  #pragma unroll
  for (int off = 16; off > 0; off >>= 1) v += __shfl_xor(v, off, 32);
  return v;
}
__device__ __forceinline__ float eluf(float x) {
  return x > 0.f ? x : __expf(x) - 1.f;
}

// Author: channels {self(z), written_by(c)}. One wave per node.
__global__ __launch_bounds__(256) void finalize_author(
    const float* __restrict__ zr, const float* __restrict__ agg,
    const float* __restrict__ deg, const float* __restrict__ bself,
    const float* __restrict__ bconv, const float* __restrict__ qvec,
    const float* __restrict__ qsc, const float* __restrict__ kvec,
    const float* __restrict__ ksc, float* __restrict__ out, int n) {
  const int node = (int)((blockIdx.x * (size_t)blockDim.x + threadIdx.x) >> 5);
  const int lane = threadIdx.x & 31;
  if (node >= n) return;
  const int d0 = lane * 4;
  const float inv = 1.f / fmaxf(deg[node], 1.f);
  float z[4], c[4];
  float hr_p = 0.f, hl_p = 0.f, hc_p = 0.f;
  #pragma unroll
  for (int t = 0; t < 4; ++t) {
    z[t] = zr[(size_t)node * NDIM + d0 + t] + bself[d0 + t];
    c[t] = agg[(size_t)node * NDIM + d0 + t] * inv + bconv[d0 + t];
    const float qv = qvec[d0 + t], kv = kvec[d0 + t];
    hr_p += z[t] * qv;
    hl_p += z[t] * kv;
    hc_p += c[t] * kv;
  }
  const float hr = wave_sum32(hr_p) + qsc[0];
  const float hl = wave_sum32(hl_p) + ksc[0];
  const float hc = wave_sum32(hc_p) + ksc[0];
  const float e0 = eluf(hl + hr);   // self
  const float e1 = eluf(hc + hr);   // written_by
  const float mx = fmaxf(e0, e1);
  const float w0 = __expf(e0 - mx), w1 = __expf(e1 - mx);
  const float is = 1.f / (w0 + w1);
  #pragma unroll
  for (int t = 0; t < 4; ++t)
    out[(size_t)node * NDIM + d0 + t] = eluf((w0 * z[t] + w1 * c[t]) * is);
}

// Paper: channels {self(z), writes(c1), cites(c2)}. One wave per node.
__global__ __launch_bounds__(256) void finalize_paper(
    const float* __restrict__ zr,
    const float* __restrict__ agg1, const float* __restrict__ deg1,
    const float* __restrict__ bc1,
    const float* __restrict__ agg2, const float* __restrict__ deg2,
    const float* __restrict__ bc2,
    const float* __restrict__ bself, const float* __restrict__ qvec,
    const float* __restrict__ qsc, const float* __restrict__ kvec,
    const float* __restrict__ ksc, float* __restrict__ out, int n) {
  const int node = (int)((blockIdx.x * (size_t)blockDim.x + threadIdx.x) >> 5);
  const int lane = threadIdx.x & 31;
  if (node >= n) return;
  const int d0 = lane * 4;
  const float inv1 = 1.f / fmaxf(deg1[node], 1.f);
  const float inv2 = 1.f / fmaxf(deg2[node], 1.f);
  float z[4], c1[4], c2[4];
  float hr_p = 0.f, hl_p = 0.f, h1_p = 0.f, h2_p = 0.f;
  #pragma unroll
  for (int t = 0; t < 4; ++t) {
    const size_t idx = (size_t)node * NDIM + d0 + t;
    z[t]  = zr[idx] + bself[d0 + t];
    c1[t] = agg1[idx] * inv1 + bc1[d0 + t];
    c2[t] = agg2[idx] * inv2 + bc2[d0 + t];
    const float qv = qvec[d0 + t], kv = kvec[d0 + t];
    hr_p += z[t] * qv;
    hl_p += z[t] * kv;
    h1_p += c1[t] * kv;
    h2_p += c2[t] * kv;
  }
  const float hr = wave_sum32(hr_p) + qsc[0];
  const float hl = wave_sum32(hl_p) + ksc[0];
  const float h1 = wave_sum32(h1_p) + ksc[0];
  const float h2 = wave_sum32(h2_p) + ksc[0];
  const float e0 = eluf(hl + hr);   // self
  const float e1 = eluf(h1 + hr);   // writes
  const float e2 = eluf(h2 + hr);   // cites
  const float mx = fmaxf(fmaxf(e0, e1), e2);
  const float w0 = __expf(e0 - mx), w1 = __expf(e1 - mx), w2 = __expf(e2 - mx);
  const float is = 1.f / (w0 + w1 + w2);
  #pragma unroll
  for (int t = 0; t < 4; ++t)
    out[(size_t)node * NDIM + d0 + t] =
        eluf((w0 * z[t] + w1 * c1[t] + w2 * c2[t]) * is);
}

// ---------------------------------------------------------------------------
// Launch
// ---------------------------------------------------------------------------
extern "C" void kernel_launch(void* const* d_in, const int* in_sizes, int n_in,
                              void* d_out, int out_size, void* d_ws,
                              size_t ws_size, hipStream_t stream) {
  const float* h_author = (const float*)d_in[0];
  const float* h_paper  = (const float*)d_in[1];
  const float* Wself_a  = (const float*)d_in[2];
  const float* bself_a  = (const float*)d_in[3];
  const float* Wself_p  = (const float*)d_in[4];
  const float* bself_p  = (const float*)d_in[5];
  const float* Wq_a = (const float*)d_in[6];  const float* bq_a = (const float*)d_in[7];
  const float* Wq_p = (const float*)d_in[8];  const float* bq_p = (const float*)d_in[9];
  const float* Wk_a = (const float*)d_in[10]; const float* bk_a = (const float*)d_in[11];
  const float* Wk_p = (const float*)d_in[12]; const float* bk_p = (const float*)d_in[13];
  const float* Wal_a = (const float*)d_in[14]; const float* bal_a = (const float*)d_in[15];
  const float* Wal_p = (const float*)d_in[16]; const float* bal_p = (const float*)d_in[17];
  const float* War_a = (const float*)d_in[18]; const float* bar_a = (const float*)d_in[19];
  const float* War_p = (const float*)d_in[20]; const float* bar_p = (const float*)d_in[21];
  const float* Wc_writes = (const float*)d_in[22]; const float* bc_writes = (const float*)d_in[23];
  const float* Wc_wb     = (const float*)d_in[24]; const float* bc_wb     = (const float*)d_in[25];
  const float* Wc_cites  = (const float*)d_in[26]; const float* bc_cites  = (const float*)d_in[27];
  const int* writes_src = (const int*)d_in[28];
  const int* writes_dst = (const int*)d_in[29];
  const int* wb_src     = (const int*)d_in[30];
  const int* wb_dst     = (const int*)d_in[31];
  const int* cites_src  = (const int*)d_in[32];
  const int* cites_dst  = (const int*)d_in[33];

  const int NA = in_sizes[0] / KDIM;   // 50000
  const int NP = in_sizes[1] / KDIM;   // 50000
  const int E  = in_sizes[28];         // 800000

  // --- workspace layout (floats) ---
  float* ws = (float*)d_ws;
  const size_t ZA = (size_t)NA * NDIM;
  const size_t ZP = (size_t)NP * NDIM;
  float* z_a        = ws;                      // [NA,128]
  float* z_p        = z_a + ZA;                // [NP,128]
  float* m_writes   = z_p + ZP;                // [NA,128]
  float* m_wb       = m_writes + ZA;           // [NP,128]
  float* m_cites    = m_wb + ZP;               // [NP,128]
  float* agg_writes = m_cites + ZP;            // [NP,128]  (zeroed)
  float* agg_wb     = agg_writes + ZP;         // [NA,128]  (zeroed)
  float* agg_cites  = agg_wb + ZA;             // [NP,128]  (zeroed)
  float* deg_writes = agg_cites + ZP;          // [NP]      (zeroed)
  float* deg_wb     = deg_writes + NP;         // [NA]      (zeroed)
  float* deg_cites  = deg_wb + NA;             // [NP]      (zeroed)
  float* vecs       = deg_cites + NP;          // 4 * (128 + 1)
  float* qvec_a = vecs;        float* qs_a = qvec_a + 128;
  float* kvec_a = qs_a + 1;    float* ks_a = kvec_a + 128;
  float* qvec_p = ks_a + 1;    float* qs_p = qvec_p + 128;
  float* kvec_p = qs_p + 1;    float* ks_p = kvec_p + 128;

  // zero the atomic accumulators + degrees (single contiguous region)
  const size_t zero_floats = (size_t)(ZP + ZA + ZP) + (size_t)(NP + NA + NP);
  hipMemsetAsync(agg_writes, 0, zero_floats * sizeof(float), stream);

  // --- fold attention linear chains into vec+scalar ---
  combine_attn<<<1, 128, 0, stream>>>(Wq_a, War_a, bq_a, bar_a, qvec_a, qs_a);
  combine_attn<<<1, 128, 0, stream>>>(Wk_a, Wal_a, bk_a, bal_a, kvec_a, ks_a);
  combine_attn<<<1, 128, 0, stream>>>(Wq_p, War_p, bq_p, bar_p, qvec_p, qs_p);
  combine_attn<<<1, 128, 0, stream>>>(Wk_p, Wal_p, bk_p, bal_p, kvec_p, ks_p);

  // --- 5 dense projections via bf16 WMMA ---
  const int tilesA = (NA + 15) / 16, tilesP = (NP + 15) / 16;
  const int gridA = (tilesA + 3) / 4, gridP = (tilesP + 3) / 4;
  gemm_bf16_wmma<<<gridA, 128, 0, stream>>>(h_author, Wself_a,   z_a,      NA);
  gemm_bf16_wmma<<<gridP, 128, 0, stream>>>(h_paper,  Wself_p,   z_p,      NP);
  gemm_bf16_wmma<<<gridA, 128, 0, stream>>>(h_author, Wc_writes, m_writes, NA);
  gemm_bf16_wmma<<<gridP, 128, 0, stream>>>(h_paper,  Wc_wb,     m_wb,     NP);
  gemm_bf16_wmma<<<gridP, 128, 0, stream>>>(h_paper,  Wc_cites,  m_cites,  NP);

  // --- per-relation message aggregation (atomics, L2-resident) ---
  const int egrid = (E + 7) / 8;  // 8 edge-waves per 256-thread block
  edge_scatter<<<egrid, 256, 0, stream>>>(m_writes, writes_src, writes_dst,
                                          agg_writes, deg_writes, E);
  edge_scatter<<<egrid, 256, 0, stream>>>(m_wb, wb_src, wb_dst,
                                          agg_wb, deg_wb, E);
  edge_scatter<<<egrid, 256, 0, stream>>>(m_cites, cites_src, cites_dst,
                                          agg_cites, deg_cites, E);

  // --- attention softmax + combine + elu ---
  float* out_a = (float*)d_out;
  float* out_p = out_a + ZA;
  const int fgridA = (NA * 32 + 255) / 256;
  const int fgridP = (NP * 32 + 255) / 256;
  finalize_author<<<fgridA, 256, 0, stream>>>(z_a, agg_wb, deg_wb, bself_a,
                                              bc_wb, qvec_a, qs_a, kvec_a,
                                              ks_a, out_a, NA);
  finalize_paper<<<fgridP, 256, 0, stream>>>(z_p, agg_writes, deg_writes,
                                             bc_writes, agg_cites, deg_cites,
                                             bc_cites, bself_p, qvec_p, qs_p,
                                             kvec_p, ks_p, out_p, NP);
}
